// LoraModel_51677046505966
// MI455X (gfx1250) — compile-verified, matrix operand order checked by reading, not run on hardware
//
#include <hip/hip_runtime.h>
#include <math.h>

// ---------------------------------------------------------------------------
// MoE-LoRA fused forward for MI455X (gfx1250, wave32, WMMA, async-to-LDS).
//   out = x @ W^T + bias + SCALING * ((gated x@loraA^T) @ loraB)
//   plus routing_weight [B,S,E] and gate_score [B,S,E].
//
// Roofline: base GEMM is 275 GFLOP over ~335 MB -> compute bound at bf16 WMMA
// rates, so operands are pre-converted to bf16 once (extra ~300 MB ~= 13 us at
// 23.3 TB/s) and the GEMM inner loop is a pure async pipeline:
//   global_load_async_to_lds_b128 (next tile)  ||  v_wmma_f32_16x16x32_bf16
// with double-buffered LDS and s_wait_asynccnt for stage handoff.
// ---------------------------------------------------------------------------

typedef __bf16 bf16_t;
typedef __attribute__((ext_vector_type(16))) __bf16 v16bf;
typedef __attribute__((ext_vector_type(8)))  __bf16 v8bf;
typedef __attribute__((ext_vector_type(8)))  float  v8f;
typedef __attribute__((ext_vector_type(4)))  int    v4i;

union frag_u { v8bf h[2]; v16bf v; };

#define NTOK   8192      // B*S
#define DIN    4096
#define DOUT   4096
#define NEXP   8
#define RANK   16
#define ER     128       // NEXP*RANK
#define SCALING 2.0f     // lora_alpha / r = 32/16

// GEMM tiling: block 128x128, K-stage 64 (2 WMMA substeps), 8 waves (2x4),
// wave tile 64x32 -> 16 v_wmma per stage per wave.
#define BM 128
#define BN 128
#define BK 64
#define LDT 72           // padded LDS row stride (bf16): 144 B, 16B-aligned
#define NK1 (DIN / BK)   // 64 base stages
#define NKT (NK1 + ER / BK) // + 2 LoRA stages = 66

// ---- CDNA5 async-to-LDS path (guarded; falls back to sync b128 copies) ----
#if __has_builtin(__builtin_amdgcn_global_load_async_to_lds_b128)
#define HAVE_ASYNC 1
#else
#define HAVE_ASYNC 0
#endif

#if __has_builtin(__builtin_amdgcn_s_wait_asynccnt)
#define WAIT_ASYNC(n) __builtin_amdgcn_s_wait_asynccnt(n)
#elif HAVE_ASYNC
#define WAIT_ASYNC(n) asm volatile("s_wait_asynccnt %0" ::"i"(n) : "memory")
#else
#define WAIT_ASYNC(n)
#endif

__device__ __forceinline__
void copy16_g2l(const bf16_t* __restrict__ g, bf16_t* __restrict__ l)
{
#if HAVE_ASYNC
    __builtin_amdgcn_global_load_async_to_lds_b128(
        (__attribute__((address_space(1))) v4i*)g,
        (__attribute__((address_space(3))) v4i*)l, 0, 0);
#else
    *(v8bf*)l = *(const v8bf*)g;
#endif
}

// ---------------------------------------------------------------------------
// Pre-pass: fp32 -> bf16 bulk convert (vectorized, streamed once).
// ---------------------------------------------------------------------------
__global__ __launch_bounds__(256)
void f32_to_bf16_kernel(const float* __restrict__ src,
                        bf16_t* __restrict__ dst, long n)
{
    long i = (((long)blockIdx.x * 256) + threadIdx.x) * 4;
    if (i >= n) return;
    float4 v = *(const float4*)(src + i);
    union { bf16_t e[4]; uint2 u; } p;
    p.e[0] = (bf16_t)v.x; p.e[1] = (bf16_t)v.y;
    p.e[2] = (bf16_t)v.z; p.e[3] = (bf16_t)v.w;
    *(uint2*)(dst + i) = p.u;
}

// Build Lb[n][e*16+r] = SCALING * lora_B[e, n, r]  (bf16, [DOUT][ER])
__global__ __launch_bounds__(256)
void build_lb_kernel(const float* __restrict__ lora_B, bf16_t* __restrict__ Lb)
{
    int i = blockIdx.x * 256 + threadIdx.x;      // 0 .. DOUT*ER-1
    int n  = i >> 7;
    int kk = i & (ER - 1);
    int e = kk >> 4, r = kk & 15;
    Lb[i] = (bf16_t)(SCALING * lora_B[((size_t)e * DOUT + n) * RANK + r]);
}

// ---------------------------------------------------------------------------
// Router: gate, softmax, top-2, renorm + gated LoRA-A projection (bf16 out).
// One 256-thread block per token; wave w owns expert w.
// ---------------------------------------------------------------------------
__global__ __launch_bounds__(256)
void lora_router_kernel(const float* __restrict__ x,
                        const float* __restrict__ route_W,
                        const float* __restrict__ lora_A,
                        float* __restrict__ rw_out,
                        float* __restrict__ gs_out,
                        bf16_t* __restrict__ hwb)
{
    __shared__ float xs[DIN];
    __shared__ float h_sh[ER];
    __shared__ float g_sh[NEXP];
    __shared__ float rw_sh[NEXP];

    const int tok  = blockIdx.x;
    const int tid  = threadIdx.x;
    const int lane = tid & 31;
    const int wave = tid >> 5;

    const float* xr = x + (size_t)tok * DIN;
    for (int i = tid; i < DIN; i += 256) xs[i] = xr[i];
    __syncthreads();

    {
        const float* wr = route_W + (size_t)wave * DIN;
        float acc = 0.f;
        for (int k = lane; k < DIN; k += 32) acc += xs[k] * wr[k];
        #pragma unroll
        for (int off = 16; off > 0; off >>= 1) acc += __shfl_xor(acc, off, 32);
        if (lane == 0) g_sh[wave] = acc;
    }
    for (int r = 0; r < RANK; ++r) {
        const float* ar = lora_A + ((size_t)wave * RANK + r) * DIN;
        float acc = 0.f;
        for (int k = lane; k < DIN; k += 32) acc += xs[k] * ar[k];
        #pragma unroll
        for (int off = 16; off > 0; off >>= 1) acc += __shfl_xor(acc, off, 32);
        if (lane == 0) h_sh[wave * RANK + r] = acc;
    }
    __syncthreads();

    if (tid == 0) {
        float g[NEXP], p[NEXP];
        float mx = -3.402823466e38f;
        for (int i = 0; i < NEXP; ++i) { g[i] = g_sh[i]; mx = fmaxf(mx, g[i]); }
        float s = 0.f;
        for (int i = 0; i < NEXP; ++i) { p[i] = __expf(g[i] - mx); s += p[i]; }
        for (int i = 0; i < NEXP; ++i) p[i] /= s;
        int i1 = 0;
        for (int i = 1; i < NEXP; ++i) if (p[i] > p[i1]) i1 = i;
        int i2 = (i1 == 0) ? 1 : 0;
        for (int i = 0; i < NEXP; ++i) if (i != i1 && p[i] > p[i2]) i2 = i;
        float t = p[i1] + p[i2] + 1e-9f;
        for (int i = 0; i < NEXP; ++i) {
            float w = (i == i1) ? p[i1] / t : ((i == i2) ? p[i2] / t : 0.f);
            rw_sh[i] = w;
            rw_out[(size_t)tok * NEXP + i] = w;
            gs_out[(size_t)tok * NEXP + i] = g[i];
        }
    }
    __syncthreads();

    if (tid < ER)
        hwb[(size_t)tok * ER + tid] = (bf16_t)(h_sh[tid] * rw_sh[tid >> 4]);
}

// ---------------------------------------------------------------------------
// Stage one 128x64 A tile + 128x64 B tile into LDS (16B chunks, async).
// 1024 chunks per tile / 256 threads = 4 async instrs per wave per tile
// -> 8 outstanding async ops per wave per stage.
// ---------------------------------------------------------------------------
__device__ __forceinline__
void stage_tiles(const bf16_t* __restrict__ gA, int sA,
                 const bf16_t* __restrict__ gB, int sB,
                 bf16_t* __restrict__ lA, bf16_t* __restrict__ lB, int tid)
{
    #pragma unroll
    for (int j = 0; j < 4; ++j) {
        int i = tid + j * 256;
        int row = i >> 3, c = (i & 7) << 3;       // 8 bf16 = 16 B chunks
        copy16_g2l(gA + (size_t)row * sA + c, lA + row * LDT + c);
    }
    #pragma unroll
    for (int j = 0; j < 4; ++j) {
        int i = tid + j * 256;
        int row = i >> 3, c = (i & 7) << 3;
        copy16_g2l(gB + (size_t)row * sB + c, lB + row * LDT + c);
    }
}

// One 32-deep WMMA substep (koff = 0 or 32) over the 128x128 block tile.
__device__ __forceinline__
void tile_mma(const bf16_t* __restrict__ As, const bf16_t* __restrict__ Bs,
              v8f (&acc)[4][2], int waveM, int waveN, int half, int l15,
              int koff)
{
    frag_u a[4], b[2];
    #pragma unroll
    for (int mt = 0; mt < 4; ++mt) {
        const bf16_t* base = &As[(waveM * 64 + mt * 16 + l15) * LDT + koff];
        a[mt].h[0] = *(const v8bf*)(base + half * 8);
        a[mt].h[1] = *(const v8bf*)(base + 16 + half * 8);
    }
    #pragma unroll
    for (int nt = 0; nt < 2; ++nt) {
        const bf16_t* base = &Bs[(waveN * 32 + nt * 16 + l15) * LDT + koff];
        b[nt].h[0] = *(const v8bf*)(base + half * 8);
        b[nt].h[1] = *(const v8bf*)(base + 16 + half * 8);
    }
    #pragma unroll
    for (int mt = 0; mt < 4; ++mt)
        #pragma unroll
        for (int nt = 0; nt < 2; ++nt)
            acc[mt][nt] = __builtin_amdgcn_wmma_f32_16x16x32_bf16(
                false, a[mt].v, false, b[nt].v, (short)0, acc[mt][nt],
                false, false);
}

// ---------------------------------------------------------------------------
// Fused GEMM: acc = bias; 64 stages of x@W^T then 2 stages of hw@(S*loraB),
// one double-buffered async pipeline across all 66 stages.
// ---------------------------------------------------------------------------
__global__ __launch_bounds__(256)
void lora_gemm_kernel(const bf16_t* __restrict__ xb,
                      const bf16_t* __restrict__ Wb,
                      const bf16_t* __restrict__ hwb,
                      const bf16_t* __restrict__ Lb,
                      const float* __restrict__ bias,
                      float* __restrict__ out)
{
    __shared__ bf16_t As[2][BM * LDT];
    __shared__ bf16_t Bs[2][BN * LDT];

    const int tid   = threadIdx.x;
    const int lane  = tid & 31;
    const int wave  = tid >> 5;
    const int waveM = wave >> 2;
    const int waveN = wave & 3;
    const int half  = lane >> 4;
    const int l15   = lane & 15;

    const int m0 = blockIdx.y * BM;
    const int n0 = blockIdx.x * BN;

    v8f acc[4][2];
    #pragma unroll
    for (int nt = 0; nt < 2; ++nt) {
        float bv = bias[n0 + waveN * 32 + nt * 16 + l15];
        #pragma unroll
        for (int mt = 0; mt < 4; ++mt)
            #pragma unroll
            for (int v = 0; v < 8; ++v) acc[mt][nt][v] = bv;
    }

    // per-stage operand selection (uniform/scalar branch)
    auto srcA = [&](int kt, int& stride) -> const bf16_t* {
        if (kt < NK1) { stride = DIN; return xb + (size_t)m0 * DIN + kt * BK; }
        stride = ER;  return hwb + (size_t)m0 * ER + (kt - NK1) * BK;
    };
    auto srcB = [&](int kt, int& stride) -> const bf16_t* {
        if (kt < NK1) { stride = DIN; return Wb + (size_t)n0 * DIN + kt * BK; }
        stride = ER;  return Lb + (size_t)n0 * ER + (kt - NK1) * BK;
    };

    // prologue: stage 0 into buffer 0
    {
        int sA, sB;
        const bf16_t* gA = srcA(0, sA);
        const bf16_t* gB = srcB(0, sB);
        stage_tiles(gA, sA, gB, sB, As[0], Bs[0], tid);
    }

    for (int kt = 0; kt < NKT; ++kt) {
        const int cur = kt & 1;
        if (kt + 1 < NKT) {
            int sA, sB;
            const bf16_t* gA = srcA(kt + 1, sA);
            const bf16_t* gB = srcB(kt + 1, sB);
            stage_tiles(gA, sA, gB, sB, As[cur ^ 1], Bs[cur ^ 1], tid);
            WAIT_ASYNC(8);      // previous stage's 8 async ops complete
        } else {
            WAIT_ASYNC(0);
        }
        __syncthreads();        // all waves' tile writes visible

        tile_mma(As[cur], Bs[cur], acc, waveM, waveN, half, l15, 0);
        tile_mma(As[cur], Bs[cur], acc, waveM, waveN, half, l15, 32);

        __syncthreads();        // done reading before buffer reuse
    }

    // streamed store, non-temporal (no reuse of the 134 MB result)
    #pragma unroll
    for (int mt = 0; mt < 4; ++mt) {
        int crow = m0 + waveM * 64 + mt * 16 + half * 8;
        #pragma unroll
        for (int nt = 0; nt < 2; ++nt) {
            int ccol = n0 + waveN * 32 + nt * 16 + l15;
            #pragma unroll
            for (int v = 0; v < 8; ++v)
                __builtin_nontemporal_store(
                    acc[mt][nt][v], out + (size_t)(crow + v) * DOUT + ccol);
        }
    }
}

// ---------------------------------------------------------------------------
extern "C" void kernel_launch(void* const* d_in, const int* in_sizes, int n_in,
                              void* d_out, int out_size, void* d_ws, size_t ws_size,
                              hipStream_t stream)
{
    (void)in_sizes; (void)n_in; (void)out_size; (void)ws_size;

    const float* x       = (const float*)d_in[0];
    const float* W       = (const float*)d_in[1];
    const float* bias    = (const float*)d_in[2];
    const float* route_W = (const float*)d_in[3];
    const float* lora_A  = (const float*)d_in[4];
    const float* lora_B  = (const float*)d_in[5];

    float* outR   = (float*)d_out;                    // [8192,4096]
    float* rw_out = outR + (size_t)NTOK * DOUT;       // [8192,8]
    float* gs_out = rw_out + (size_t)NTOK * NEXP;     // [8192,8]

    // workspace layout (bf16 operands)
    char* ws = (char*)d_ws;
    bf16_t* xb  = (bf16_t*)ws;                        ws += (size_t)NTOK * DIN * 2;
    bf16_t* Wb  = (bf16_t*)ws;                        ws += (size_t)DOUT * DIN * 2;
    bf16_t* Lb  = (bf16_t*)ws;                        ws += (size_t)DOUT * ER * 2;
    bf16_t* hwb = (bf16_t*)ws;

    // pre-pass conversions (independent; all on stream)
    f32_to_bf16_kernel<<<(NTOK * (long)DIN) / 1024, 256, 0, stream>>>(
        x, xb, (long)NTOK * DIN);
    f32_to_bf16_kernel<<<(DOUT * (long)DIN) / 1024, 256, 0, stream>>>(
        W, Wb, (long)DOUT * DIN);
    build_lb_kernel<<<(DOUT * ER) / 256, 256, 0, stream>>>(lora_B, Lb);

    lora_router_kernel<<<NTOK, 256, 0, stream>>>(x, route_W, lora_A,
                                                 rw_out, gs_out, hwb);

    dim3 grid(DOUT / BN, NTOK / BM);                  // 32 x 64
    lora_gemm_kernel<<<grid, 256, 0, stream>>>(xb, Wb, hwb, Lb, bias, outR);
}
